// AttentionMV_78159814852924
// MI455X (gfx1250) — matrix operand order repaired and checked by reading, not compiled, not run on hardware
//
#include <hip/hip_runtime.h>
#include <hip/hip_bf16.h>
#include <math.h>

#define BB 32
#define TT 2048
#define EE 1024

#define TM 128
#define TN 128
#define TK 32
#define LDA 56   // padded k-stride (bf16 units): 112B rows -> 16B aligned, conflict-free

typedef __bf16 bf16;
typedef __attribute__((ext_vector_type(4)))  __bf16 bf16x4;
typedef __attribute__((ext_vector_type(16))) __bf16 v16bf;
typedef __attribute__((ext_vector_type(8)))  float  v8f;

#if __has_builtin(__builtin_amdgcn_tanhf)
#define FAST_TANH(x) __builtin_amdgcn_tanhf(x)
#elif __has_builtin(__builtin_amdgcn_tanh_f32)
#define FAST_TANH(x) __builtin_amdgcn_tanh_f32(x)
#else
#define FAST_TANH(x) tanhf(x)
#endif

union V16 { uint4 u[2]; v16bf v; };
static __device__ inline v16bf pack16(uint4 a, uint4 b) { V16 t; t.u[0] = a; t.u[1] = b; return t.v; }
union BF2 { unsigned int u; __bf16 h[2]; };

struct LdsIn { bf16 A[TM][LDA]; bf16 Bm[TN][LDA]; };
union  LdsU  { LdsIn in[2]; bf16 out[TM][TN]; };   // 2x28KB ping-pong vs 32KB epilogue stage

// Stage one K-tile: enc[.,k0:k0+32] -> A (bf16), W[k0:k0+32,.] -> Bm transposed [n][k].
static __device__ inline void stage_tile(LdsIn& buf, const float* __restrict__ encBase,
                                         const float* __restrict__ Wbase, int k0, int tid)
{
    // A: 128 rows x 32 k. 8 threads/row, 32 rows/pass x 4 passes, packed 8B stores.
    {
        const int lr = tid & 7;
        const int r0 = tid >> 3;
        #pragma unroll
        for (int rr = 0; rr < 4; ++rr) {
            const int r = r0 + rr * 32;
            const float4 v = *(const float4*)(encBase + (size_t)r * EE + k0 + lr * 4);
            bf16x4 p; p.x = (bf16)v.x; p.y = (bf16)v.y; p.z = (bf16)v.z; p.w = (bf16)v.w;
            *(bf16x4*)&buf.A[r][lr * 4] = p;
        }
    }
    // B: 32 k-rows x 128 n. Each thread: 4-col group (lc) x 4 consecutive k (kq),
    // register transpose -> 4x packed 8B ds stores along k.
    {
        const int lc = tid & 31;     // n-group of 4
        const int kq = tid >> 5;     // k-quad: rows kq*4..kq*4+3
        float4 v[4];
        #pragma unroll
        for (int j = 0; j < 4; ++j)
            v[j] = *(const float4*)(Wbase + (size_t)(k0 + kq * 4 + j) * EE + lc * 4);
        #pragma unroll
        for (int j = 0; j < 4; ++j) {
            bf16x4 p;
            p.x = (bf16)((j == 0) ? v[0].x : (j == 1) ? v[0].y : (j == 2) ? v[0].z : v[0].w);
            p.y = (bf16)((j == 0) ? v[1].x : (j == 1) ? v[1].y : (j == 2) ? v[1].z : v[1].w);
            p.z = (bf16)((j == 0) ? v[2].x : (j == 1) ? v[2].y : (j == 2) ? v[2].z : v[2].w);
            p.w = (bf16)((j == 0) ? v[3].x : (j == 1) ? v[3].y : (j == 2) ? v[3].z : v[3].w);
            *(bf16x4*)&buf.Bm[lc * 4 + j][kq * 4] = p;
        }
    }
}

// ---------------------------------------------------------------------------
// Kernel 1: ht[b, t0:t0+128, n0:n0+128] = tanh(enc @ W + bias), bf16 output.
// 256 threads = 8 waves; wave grid 2(M) x 4(N); each wave: 4x2 WMMA tiles.
// Double-buffered LDS: next tile's loads issued before current tile's WMMAs.
// ---------------------------------------------------------------------------
__global__ __launch_bounds__(256) void att_gemm_tanh(
    const float* __restrict__ enc, const float* __restrict__ W,
    const float* __restrict__ bias, bf16* __restrict__ ht)
{
    __shared__ LdsU lds;

    const int tid  = threadIdx.x;
    const int b    = blockIdx.z;
    const int t0   = blockIdx.y * TM;
    const int n0   = blockIdx.x * TN;
    const int wave = tid >> 5;
    const int lane = tid & 31;
    const int wm   = wave >> 2;       // 0..1 (M)
    const int wn   = wave & 3;        // 0..3 (N)
    const int l15  = lane & 15;
    const int hi   = (lane >> 4) & 1; // lane-group select

    v8f acc[4][2] = {};

    const float* encBase = enc + ((size_t)b * TT + t0) * EE;
    const float* Wbase   = W + n0;

    constexpr int NK = EE / TK;       // 32 K-steps
    int cur = 0;
    stage_tile(lds.in[0], encBase, Wbase, 0, tid);

    for (int kk = 0; kk < NK; ++kk) {
        __syncthreads();                               // buf[cur] ready, buf[cur^1] free
        if (kk + 1 < NK)
            stage_tile(lds.in[cur ^ 1], encBase, Wbase, (kk + 1) * TK, tid);

        // ---- fragments per ISA 16-bit layouts ------------------------------
        v16bf af[4];
        #pragma unroll
        for (int mi = 0; mi < 4; ++mi) {
            const bf16* rp = &lds.in[cur].A[wm * 64 + mi * 16 + l15][hi * 8];
            af[mi] = pack16(*(const uint4*)(rp), *(const uint4*)(rp + 16));
        }
        v16bf bfr[2];
        #pragma unroll
        for (int ni = 0; ni < 2; ++ni) {
            const bf16* cp = &lds.in[cur].Bm[wn * 32 + ni * 16 + l15][hi * 8];
            bfr[ni] = pack16(*(const uint4*)(cp), *(const uint4*)(cp + 16));
        }
        #pragma unroll
        for (int mi = 0; mi < 4; ++mi)
            #pragma unroll
            for (int ni = 0; ni < 2; ++ni)
                acc[mi][ni] = __builtin_amdgcn_wmma_f32_16x16x32_bf16(
                    false, af[mi], false, bfr[ni], (short)0, acc[mi][ni], false, false);
        cur ^= 1;
    }

    // ---- epilogue: +bias, hw tanh, stage to LDS, coalesced bf16 store ------
    float biasv[2];
    #pragma unroll
    for (int ni = 0; ni < 2; ++ni)
        biasv[ni] = bias[n0 + wn * 32 + ni * 16 + l15];

    __syncthreads();                                   // done reading lds.in
    #pragma unroll
    for (int mi = 0; mi < 4; ++mi)
        #pragma unroll
        for (int ni = 0; ni < 2; ++ni)
            #pragma unroll
            for (int v = 0; v < 8; ++v) {
                const float h = FAST_TANH(acc[mi][ni][v] + biasv[ni]);
                const int row = wm * 64 + mi * 16 + hi * 8 + v;
                const int col = wn * 32 + ni * 16 + l15;
                lds.out[row][col] = (bf16)h;
            }
    __syncthreads();

    {
        const int r    = tid >> 1;
        const int cgrp = (tid & 1) * 64;               // 64 bf16 = 128 B per thread
        bf16* dst = ht + ((size_t)b * TT + t0 + r) * EE + n0 + cgrp;
        const uint4* src = (const uint4*)&lds.out[r][cgrp];
        #pragma unroll
        for (int i = 0; i < 8; ++i)
            *(uint4*)(dst + i * 8) = src[i];
    }
}

// ---------------------------------------------------------------------------
// Kernel 2: scores[b,t] = ctx[b] . ht[b,t,:]   (warp per row)
// ---------------------------------------------------------------------------
__global__ __launch_bounds__(256) void att_scores(
    const bf16* __restrict__ ht, const float* __restrict__ ctx,
    float* __restrict__ scores)
{
    const int warp = threadIdx.x >> 5;
    const int lane = threadIdx.x & 31;
    const int row  = blockIdx.x * 8 + warp;    // over B*T
    const int b    = row >> 11;                // T = 2048

    const bf16*  hp = ht + (size_t)row * EE;
    const float* cp = ctx + (size_t)b * EE;

    float sum = 0.f;
    #pragma unroll 4
    for (int i = 0; i < EE / 64; ++i) {
        const int e = i * 64 + lane * 2;
        BF2 u; u.u = *(const unsigned int*)(hp + e);
        const float2 c = *(const float2*)(cp + e);
        sum += (float)u.h[0] * c.x + (float)u.h[1] * c.y;
    }
    #pragma unroll
    for (int off = 16; off > 0; off >>= 1)
        sum += __shfl_xor(sum, off, 32);
    if (lane == 0) scores[row] = sum;
}

// ---------------------------------------------------------------------------
// Kernel 3: at[b,:] = softmax(scores[b,:]) over T=2048 (one block per b)
// ---------------------------------------------------------------------------
__global__ __launch_bounds__(256) void att_softmax(
    const float* __restrict__ scores, float* __restrict__ at)
{
    __shared__ float red[8];
    __shared__ float bcast[2];
    const int b = blockIdx.x, tid = threadIdx.x;
    const float* s = scores + (size_t)b * TT;

    float vals[8];
    float m = -1e30f;
    #pragma unroll
    for (int i = 0; i < 8; ++i) { vals[i] = s[tid + i * 256]; m = fmaxf(m, vals[i]); }
    #pragma unroll
    for (int off = 16; off > 0; off >>= 1) m = fmaxf(m, __shfl_xor(m, off, 32));
    if ((tid & 31) == 0) red[tid >> 5] = m;
    __syncthreads();
    if (tid == 0) {
        float mm = red[0];
        for (int i = 1; i < 8; ++i) mm = fmaxf(mm, red[i]);
        bcast[0] = mm;
    }
    __syncthreads();
    m = bcast[0];

    float lsum = 0.f;
    #pragma unroll
    for (int i = 0; i < 8; ++i) { vals[i] = __expf(vals[i] - m); lsum += vals[i]; }
    #pragma unroll
    for (int off = 16; off > 0; off >>= 1) lsum += __shfl_xor(lsum, off, 32);
    if ((tid & 31) == 0) red[tid >> 5] = lsum;
    __syncthreads();
    if (tid == 0) {
        float ss = 0.f;
        for (int i = 0; i < 8; ++i) ss += red[i];
        bcast[1] = 1.f / ss;
    }
    __syncthreads();
    const float inv = bcast[1];
    #pragma unroll
    for (int i = 0; i < 8; ++i) at[(size_t)b * TT + tid + i * 256] = vals[i] * inv;
}

// ---------------------------------------------------------------------------
// Kernel 4: partial[b, tc, e] = sum over 256 t of at[b,t]*ht[b,t,e]
// grid (E/512, B, T/256), 256 threads, 2 cols/thread (coalesced 128B/warp)
// ---------------------------------------------------------------------------
__global__ __launch_bounds__(256) void att_pool_partial(
    const bf16* __restrict__ ht, const float* __restrict__ at,
    float* __restrict__ partial)
{
    const int b  = blockIdx.y;
    const int tc = blockIdx.z;
    const int e0 = blockIdx.x * 512 + threadIdx.x * 2;

    const bf16*  hp = ht + ((size_t)b * TT + (size_t)tc * 256) * EE + e0;
    const float* ap = at + (size_t)b * TT + (size_t)tc * 256;

    float s0 = 0.f, s1 = 0.f;
    #pragma unroll 4
    for (int t = 0; t < 256; ++t) {
        const float a = ap[t];
        BF2 u; u.u = *(const unsigned int*)(hp + (size_t)t * EE);
        s0 += a * (float)u.h[0];
        s1 += a * (float)u.h[1];
    }
    float* pp = partial + ((size_t)b * 8 + tc) * EE + e0;
    pp[0] = s0; pp[1] = s1;
}

// Kernel 5: out[b,e] = sum over 8 chunks of partial
__global__ __launch_bounds__(256) void att_pool_reduce(
    const float* __restrict__ partial, float* __restrict__ out)
{
    const int idx = blockIdx.x * 256 + threadIdx.x;   // over B*E
    const int b = idx >> 10, e = idx & 1023;
    float s = 0.f;
    #pragma unroll
    for (int c = 0; c < 8; ++c) s += partial[((size_t)b * 8 + c) * EE + e];
    out[idx] = s;
}

// ---------------------------------------------------------------------------
extern "C" void kernel_launch(void* const* d_in, const int* in_sizes, int n_in,
                              void* d_out, int out_size, void* d_ws, size_t ws_size,
                              hipStream_t stream) {
    const float* enc  = (const float*)d_in[0];
    const float* ctx  = (const float*)d_in[1];
    const float* W    = (const float*)d_in[2];
    const float* bias = (const float*)d_in[3];
    float* out = (float*)d_out;

    char* ws = (char*)d_ws;
    bf16*  ht      = (bf16*)ws;                                   // B*T*E bf16 = 128 MiB
    float* scores  = (float*)(ws + (size_t)BB * TT * EE * 2);     // B*T f32
    float* at      = scores + (size_t)BB * TT;                    // B*T f32
    float* partial = at + (size_t)BB * TT;                        // B*8*E f32

    att_gemm_tanh<<<dim3(EE / TN, TT / TM, BB), 256, 0, stream>>>(enc, W, bias, ht);
    att_scores<<<dim3(BB * TT / 8), 256, 0, stream>>>(ht, ctx, scores);
    att_softmax<<<dim3(BB), 256, 0, stream>>>(scores, at);
    att_pool_partial<<<dim3(EE / 512, BB, TT / 256), 256, 0, stream>>>(ht, at, partial);
    att_pool_reduce<<<dim3(BB * EE / 256), 256, 0, stream>>>(partial, out);
}
